// SelfAttention_65481071396288
// MI455X (gfx1250) — matrix operand order repaired and checked by reading, not compiled, not run on hardware
//
#include <hip/hip_runtime.h>

// ---------------------------------------------------------------------------
// Channel self-attention (B=32, C=768, N=1024) for MI455X / gfx1250.
// All four GEMMs run on v_wmma_f32_16x16x32_f16 (f32 accumulate), arranged as
// NT GEMMs (A row-major [M,K], Bt row-major [N,K]) so every fragment load is a
// K-contiguous 16B global_load_b128 matching the CDNA5 WMMA VGPR layouts.
// Wave tile 64x64 (4x4 fragments): 16 v_wmma per 16 b128 loads per K=32 step.
// ---------------------------------------------------------------------------

typedef _Float16 h8  __attribute__((ext_vector_type(8)));
typedef _Float16 h16 __attribute__((ext_vector_type(16)));
typedef float    f8  __attribute__((ext_vector_type(8)));

constexpr int EP_QKV       = 0;  // f16 rows for q,k; transposed f16 store for v rows
constexpr int EP_SCALE_F32 = 1;  // f32 out, scaled (attention scores)
constexpr int EP_TRANS_F16 = 2;  // f16 transposed out (O^T)
constexpr int EP_BIAS_F32  = 3;  // f32 out + bias (final projection)

// CDNA5 16-bit A fragment (16x32, MxK): lane L holds row m=L%16;
//   L<16 : elems 0..7 -> K=kb+0..7,  elems 8..15 -> K=kb+16..23
//   L>=16: elems 0..7 -> K=kb+8..15, elems 8..15 -> K=kb+24..31
__device__ __forceinline__ h16 load_a_frag(const _Float16* __restrict__ row, int kb, int hi) {
    h8 lo = *(const h8*)(row + kb + 8 * hi);
    h8 up = *(const h8*)(row + kb + 16 + 8 * hi);
    return __builtin_shufflevector(lo, up, 0,1,2,3,4,5,6,7,8,9,10,11,12,13,14,15);
}

// CDNA5 16-bit B fragment (32x16, KxN): lane L holds column n=L%16;
//   L<16 : elems 0..7 -> K=kb+0..7,  elems 8..15 -> K=kb+16..23
//   L>=16: elems 0..7 -> K=kb+8..15, elems 8..15 -> K=kb+24..31
__device__ __forceinline__ h16 load_b_frag(const _Float16* __restrict__ row, int kb, int hi) {
    h8 lo = *(const h8*)(row + kb + 8 * hi);
    h8 up = *(const h8*)(row + kb + 16 + 8 * hi);
    return __builtin_shufflevector(lo, up, 0,1,2,3,4,5,6,7,8,9,10,11,12,13,14,15);
}

// D = A * Bt^T.  A: [M,K] row-major (lda), Bt: [N,K] row-major (ldb).
// Block tile 128x256: 8 waves in a 2(M) x 4(N) grid, each wave 64x64
// (4x4 WMMA fragments). All problem dims are exact multiples of the tiles ->
// no bounds checks, EXEC stays all-ones around every v_wmma (ISA requirement).
template <int MODE>
__global__ __launch_bounds__(256) void gemm_nt_wmma(
    const _Float16* __restrict__ A,  long sA, int lda,
    const _Float16* __restrict__ Bt, long sB, int ldb,
    int K,
    _Float16* __restrict__ O16, long sO16, int ldo16, int mSplit,
    _Float16* __restrict__ OT,  long sOT,  int ldoT,
    float*    __restrict__ O32, long sO32, int ldo32,
    const float* __restrict__ bias, float scale)
{
    const int b = blockIdx.z;
    A  += (long)b * sA;
    Bt += (long)b * sB;

    const int tid  = threadIdx.x;
    const int wave = tid >> 5;
    const int lane = tid & 31;
    const int hi   = lane >> 4;   // which 16-lane half
    const int l16  = lane & 15;

    const int wm = blockIdx.x * 128 + (wave >> 2) * 64;   // wave M origin (4 tiles)
    const int wn = blockIdx.y * 256 + (wave & 3)  * 64;   // wave N origin (4 tiles)

    f8 acc[4][4] = {};

    const _Float16* arow[4];
    const _Float16* brow[4];
#pragma unroll
    for (int mi = 0; mi < 4; ++mi) arow[mi] = A  + (long)(wm + mi * 16 + l16) * lda;
#pragma unroll
    for (int ni = 0; ni < 4; ++ni) brow[ni] = Bt + (long)(wn + ni * 16 + l16) * ldb;

    for (int kb = 0; kb < K; kb += 32) {
        h16 af[4], bf[4];
#pragma unroll
        for (int mi = 0; mi < 4; ++mi) af[mi] = load_a_frag(arow[mi], kb, hi);
#pragma unroll
        for (int ni = 0; ni < 4; ++ni) bf[ni] = load_b_frag(brow[ni], kb, hi);
#pragma unroll
        for (int mi = 0; mi < 4; ++mi)
#pragma unroll
            for (int ni = 0; ni < 4; ++ni)
                acc[mi][ni] = __builtin_amdgcn_wmma_f32_16x16x32_f16(
                    false, af[mi], false, bf[ni], (short)0, acc[mi][ni], false, false);
    }

    // C/D layout: lane holds (M = r + 8*hi, N = l16), r = 0..7 over the 8 accum VGPRs.
#pragma unroll
    for (int mi = 0; mi < 4; ++mi) {
#pragma unroll
        for (int ni = 0; ni < 4; ++ni) {
            const int gn  = wn + ni * 16 + l16;
            const int gm0 = wm + mi * 16 + hi * 8;
            if (MODE == EP_QKV) {
                if (gm0 < mSplit) {               // q,k rows: [d,n] f16  (uniform: split on block boundary)
                    _Float16* o = O16 + (long)b * sO16;
#pragma unroll
                    for (int r = 0; r < 8; ++r)
                        o[(long)(gm0 + r) * ldo16 + gn] = (_Float16)acc[mi][ni][r];
                } else {                          // v rows: store transposed vT[n,d]
                    h8 hv;
#pragma unroll
                    for (int r = 0; r < 8; ++r) hv[r] = (_Float16)acc[mi][ni][r];
                    *(h8*)(OT + (long)b * sOT + (long)gn * ldoT + (gm0 - mSplit)) = hv;
                }
            } else if (MODE == EP_SCALE_F32) {
                float* o = O32 + (long)b * sO32;
#pragma unroll
                for (int r = 0; r < 8; ++r)
                    o[(long)(gm0 + r) * ldo32 + gn] = acc[mi][ni][r] * scale;
            } else if (MODE == EP_TRANS_F16) {
                h8 hv;
#pragma unroll
                for (int r = 0; r < 8; ++r) hv[r] = (_Float16)acc[mi][ni][r];
                *(h8*)(OT + (long)b * sOT + (long)gn * ldoT + gm0) = hv;
            } else { // EP_BIAS_F32
                float* o = O32 + (long)b * sO32;
#pragma unroll
                for (int r = 0; r < 8; ++r)
                    o[(long)(gm0 + r) * ldo32 + gn] = acc[mi][ni][r] + bias[gm0 + r];
            }
        }
    }
}

// Row softmax over 768 cols, one wave32 per row (24 elems/lane), f32 in, f16 out.
__global__ __launch_bounds__(256) void softmax768(const float* __restrict__ S,
                                                  _Float16* __restrict__ P)
{
    const int row  = blockIdx.x * 8 + (threadIdx.x >> 5);
    const int lane = threadIdx.x & 31;
    const float* sr = S + (long)row * 768;

    float v[24];
    float m = -3.4e38f;
#pragma unroll
    for (int i = 0; i < 24; ++i) { v[i] = sr[lane + i * 32]; m = fmaxf(m, v[i]); }
#pragma unroll
    for (int off = 16; off > 0; off >>= 1) m = fmaxf(m, __shfl_xor(m, off, 32));
    float s = 0.f;
#pragma unroll
    for (int i = 0; i < 24; ++i) { v[i] = __expf(v[i] - m); s += v[i]; }
#pragma unroll
    for (int off = 16; off > 0; off >>= 1) s += __shfl_xor(s, off, 32);
    const float inv = 1.0f / s;

    _Float16* pr = P + (long)row * 768;
#pragma unroll
    for (int i = 0; i < 24; ++i) pr[lane + i * 32] = (_Float16)(v[i] * inv);
}

// x[b,c,n] f32 -> xT[b,n,c] f16 via 32x32 LDS tile (coalesced both sides).
__global__ __launch_bounds__(256) void transpose_x_f16(const float* __restrict__ x,
                                                       _Float16* __restrict__ xT)
{
    __shared__ float tile[32][33];
    const int b  = blockIdx.z;
    const int c0 = blockIdx.x * 32;
    const int n0 = blockIdx.y * 32;
    const int tx = threadIdx.x & 31;
    const int ty = threadIdx.x >> 5;

    const float* xb = x + (long)b * (768L * 1024);
#pragma unroll
    for (int i = 0; i < 32; i += 8)
        tile[ty + i][tx] = xb[(long)(c0 + ty + i) * 1024 + n0 + tx];
    __syncthreads();
    _Float16* xTb = xT + (long)b * (1024L * 768);
#pragma unroll
    for (int i = 0; i < 32; i += 8)
        xTb[(long)(n0 + ty + i) * 768 + c0 + tx] = (_Float16)tile[tx][ty + i];
}

__global__ void f32_to_f16(const float* __restrict__ in, _Float16* __restrict__ out, int n)
{
    int i = blockIdx.x * blockDim.x + threadIdx.x;
    if (i < n) out[i] = (_Float16)in[i];
}

extern "C" void kernel_launch(void* const* d_in, const int* in_sizes, int n_in,
                              void* d_out, int out_size, void* d_ws, size_t ws_size,
                              hipStream_t stream)
{
    const float* x    = (const float*)d_in[0];  // [32,768,32,32]
    const float* wqkv = (const float*)d_in[1];  // [2304,768]
    const float* wout = (const float*)d_in[2];  // [768,768]
    const float* bout = (const float*)d_in[3];  // [768]
    float* y = (float*)d_out;                   // [32,768,32,32]

    char* ws = (char*)d_ws;
    size_t off = 0;
    auto alloc = [&](size_t bytes) -> char* {
        char* p = ws + off;
        off = (off + bytes + 255) & ~(size_t)255;
        return p;
    };

    _Float16* xT    = (_Float16*)alloc(32L * 1024 * 768 * 2);   // x^T f16
    _Float16* wq16  = (_Float16*)alloc(2304L * 768 * 2);
    _Float16* wo16  = (_Float16*)alloc(768L * 768 * 2);
    _Float16* qkv16 = (_Float16*)alloc(32L * 1536 * 1024 * 2);  // q,k rows only
    _Float16* vT    = (_Float16*)alloc(32L * 1024 * 768 * 2);   // v^T f16
    float*    S     = (float*)   alloc(32L * 768 * 768 * 4);    // attn scores f32
    // Safe aliases (source buffer is dead before alias is written, every call):
    _Float16* P  = xT;            // softmax probs f16 (xT dead after GEMM1)
    _Float16* OT = (_Float16*)S;  // O^T f16          (S  dead after softmax)

    // 1) precision/layout prep
    transpose_x_f16<<<dim3(24, 32, 32), 256, 0, stream>>>(x, xT);
    f32_to_f16<<<(2304 * 768 + 255) / 256, 256, 0, stream>>>(wqkv, wq16, 2304 * 768);
    f32_to_f16<<<(768 * 768 + 255) / 256, 256, 0, stream>>>(wout, wo16, 768 * 768);

    // 2) qkv[d,n] = w_qkv[d,:] . x[:,n]   (M=2304, N=1024, K=768); v rows stored transposed
    gemm_nt_wmma<EP_QKV><<<dim3(2304 / 128, 1024 / 256, 32), 256, 0, stream>>>(
        wq16, 0, 768,
        xT, 1024L * 768, 768,
        768,
        qkv16, 1536L * 1024, 1024, 1536,
        vT, 1024L * 768, 768,
        nullptr, 0, 0,
        nullptr, 1.0f);

    // 3) S[c,d] = SCALE * q[c,:] . k[d,:]   (M=N=768, K=1024)
    gemm_nt_wmma<EP_SCALE_F32><<<dim3(768 / 128, 768 / 256, 32), 256, 0, stream>>>(
        qkv16, 1536L * 1024, 1024,
        qkv16 + 768L * 1024, 1536L * 1024, 1024,
        1024,
        nullptr, 0, 0, 0,
        nullptr, 0, 0,
        S, 768L * 768, 768,
        nullptr, 0.07216878364870322f);  // (768/4)^-0.5

    // 4) row softmax -> P f16
    softmax768<<<32 * 768 / 8, 256, 0, stream>>>(S, P);

    // 5) O[c,n] = P[c,:] . vT[n,:]  stored as O^T[n,c]   (M=768, N=1024, K=768)
    gemm_nt_wmma<EP_TRANS_F16><<<dim3(768 / 128, 1024 / 256, 32), 256, 0, stream>>>(
        P, 768L * 768, 768,
        vT, 1024L * 768, 768,
        768,
        nullptr, 0, 0, 0,
        OT, 1024L * 768, 768,
        nullptr, 0, 0,
        nullptr, 1.0f);

    // 6) y[e,n] = w_out[e,:] . OT[n,:] + b[e]   (M=768, N=1024, K=768), f32 out
    gemm_nt_wmma<EP_BIAS_F32><<<dim3(768 / 128, 1024 / 256, 32), 256, 0, stream>>>(
        wo16, 0, 768,
        OT, 1024L * 768, 768,
        768,
        nullptr, 0, 0, 0,
        nullptr, 0, 0,
        y, 768L * 1024, 1024,
        bout, 1.0f);

    (void)in_sizes; (void)n_in; (void)out_size; (void)ws_size;
}
